// MultiHeadAttention_73976516706304
// MI455X (gfx1250) — compile-verified, matrix operand order checked by reading, not run on hardware
//
#include <hip/hip_runtime.h>
#include <hip/hip_bf16.h>
#include <stdint.h>

#define D_MODEL 1024
#define NHEAD   16
#define HDIM    64
#define SEQ     2048
#define BATCH   2
#define NROWS   (BATCH*SEQ)

typedef __bf16 bf16_t;
typedef __attribute__((ext_vector_type(16))) __bf16 v16bf;
typedef __attribute__((ext_vector_type(8)))  __bf16 v8bf;
typedef __attribute__((ext_vector_type(8)))  float  v8f;
typedef __attribute__((ext_vector_type(4)))  int    v4i;

// ---------------------------------------------------------------------------
// gfx1250 async global->LDS copy (ASYNCcnt path), with safe fallback.
// Builtin signature (from hipcc diagnostic): (v4i AS1*, v4i AS3*, Iimm, Iimm)
// ---------------------------------------------------------------------------
#if defined(__gfx1250__) && __has_builtin(__builtin_amdgcn_global_load_async_to_lds_b128)
#define HAVE_ASYNC_LDS 1
#else
#define HAVE_ASYNC_LDS 0
#endif

typedef __attribute__((address_space(1))) v4i gv4i_t;
typedef __attribute__((address_space(3))) v4i lv4i_t;

static __device__ __forceinline__ void cp_async16(void* lds_dst, const void* g_src) {
#if HAVE_ASYNC_LDS
  __builtin_amdgcn_global_load_async_to_lds_b128(
      (gv4i_t*)(unsigned long long)(uintptr_t)g_src,
      (lv4i_t*)(unsigned int)(uintptr_t)lds_dst,
      /*imm offset=*/0, /*cpol=*/0);
#else
  *(v8bf*)lds_dst = *(const v8bf*)g_src;   // global_load_b128 + ds_store_b128
#endif
}

template <int N>
static __device__ __forceinline__ void async_wait() {
#if HAVE_ASYNC_LDS
#if __has_builtin(__builtin_amdgcn_s_wait_asynccnt)
  __builtin_amdgcn_s_wait_asynccnt(N);
#else
  asm volatile("s_wait_asynccnt %0" :: "n"(N) : "memory");
#endif
#endif
}

// ---------------------------------------------------------------------------
// WMMA helpers
// ---------------------------------------------------------------------------
// 16-bit A/B fragment from row-major [row][k] with leading dim `ld`:
// lane l (m/n = l%16, half = l/16) holds k = 16*(e/8) + 8*half + e%8
// -> two contiguous 16-byte runs -> two ds_load_b128.
static __device__ __forceinline__ v16bf load_frag(const bf16_t* base, int row, int ld,
                                                  int coff, int half) {
  union { v16bf v; v8bf h[2]; } u;
  const bf16_t* p = base + row * ld + coff + 8 * half;
  u.h[0] = *reinterpret_cast<const v8bf*>(p);
  u.h[1] = *reinterpret_cast<const v8bf*>(p + 16);
  return u.v;
}

static __device__ __forceinline__ v8f wmma_bf16(v16bf a, v16bf b, v8f c) {
  return __builtin_amdgcn_wmma_f32_16x16x32_bf16(
      false, a, false, b, (short)0, c, false, false);
}

// ---------------------------------------------------------------------------
// Prep kernels: one-time fp32 -> bf16 conversion (and transposition of weights
// to [n][k] so every GEMM tile is a raw, per-lane-contiguous byte copy).
// ---------------------------------------------------------------------------
__global__ __launch_bounds__(256)
void mha_cvt_bf16_kernel(const float* __restrict__ src, bf16_t* __restrict__ dst, int n) {
  int i = (blockIdx.x * 256 + threadIdx.x) * 4;
  if (i < n) {
    float4 f = *reinterpret_cast<const float4*>(src + i);
    union { bf16_t h[4]; unsigned long long u; } pk;
    pk.h[0] = (bf16_t)f.x; pk.h[1] = (bf16_t)f.y;
    pk.h[2] = (bf16_t)f.z; pk.h[3] = (bf16_t)f.w;
    *reinterpret_cast<unsigned long long*>(dst + i) = pk.u;
  }
}

__global__ __launch_bounds__(256)
void mha_cvt_transpose_kernel(const float* __restrict__ src /*[K][N]*/,
                              bf16_t* __restrict__ dst /*[N][K]*/, int K, int N) {
  int idx = blockIdx.x * 256 + threadIdx.x;
  if (idx < K * N) {
    int k = idx / N, n = idx % N;               // coalesced read along n
    dst[(size_t)n * K + k] = (bf16_t)src[idx];
  }
}

// ---------------------------------------------------------------------------
// Kernel 1: qkv = xb @ wqT^T (bf16 WMMA, f32 acc), heads split, RoPE on q/k.
// Block tile 128(M) x 64(N), K-step 32, double-buffered async LDS staging.
// V is written pre-transposed [b,h,d,s] for the attention PV operand.
// ---------------------------------------------------------------------------
__global__ __launch_bounds__(256)
void mha_qkv_rope_kernel(const bf16_t* __restrict__ xb,   // [NROWS][D_MODEL]
                         const bf16_t* __restrict__ wqT,  // [3*D_MODEL][D_MODEL]
                         bf16_t* __restrict__ qbuf,       // [B,H,S,HDIM]
                         bf16_t* __restrict__ kbuf,       // [B,H,S,HDIM]
                         bf16_t* __restrict__ vtbuf) {    // [B,H,HDIM,S]
  __shared__ alignas(16) bf16_t Asm[2][128 * 32];   // [m][k], ld=32
  __shared__ alignas(16) bf16_t Bsm[2][64 * 32];    // [n][k], ld=32

  const int tid  = threadIdx.x;
  const int wave = tid >> 5;
  const int lane = tid & 31;
  const int half = lane >> 4;
  const int ln   = lane & 15;
  const int row0 = blockIdx.x * 128;
  const int col0 = blockIdx.y * 64;

  auto issue = [&](int k0, int buf) {
    #pragma unroll
    for (int c = 0; c < 2; ++c) {                      // A: 8KB -> 2x16B/thread
      int e = (c * 256 + tid) * 8;
      int r = e >> 5, cc = e & 31;
      cp_async16(&Asm[buf][e], xb + (size_t)(row0 + r) * D_MODEL + k0 + cc);
    }
    int e = tid * 8;                                   // B: 4KB -> 1x16B/thread
    int rn = e >> 5, ck = e & 31;
    cp_async16(&Bsm[buf][e], wqT + (size_t)(col0 + rn) * D_MODEL + k0 + ck);
  };

  v8f acc[4] = {};
  const int T = D_MODEL / 32;
  issue(0, 0);
  for (int t = 0; t < T; ++t) {
    int buf = t & 1;
    if (t + 1 < T) { issue((t + 1) * 32, buf ^ 1); async_wait<3>(); }
    else           { async_wait<0>(); }
    __syncthreads();

    v16bf af = load_frag(Asm[buf], wave * 16 + ln, 32, 0, half);
    #pragma unroll
    for (int g = 0; g < 4; ++g)
      acc[g] = wmma_bf16(af, load_frag(Bsm[buf], g * 16 + ln, 32, 0, half), acc[g]);
    __syncthreads();
  }

  // Epilogue: RoPE pairs (d, d+32) sit in accumulator groups (g, g+2).
  const int part = col0 / D_MODEL;            // 0=q, 1=k, 2=v
  const int h    = (col0 % D_MODEL) / HDIM;
  const float LN10000 = 9.2103403719761836f;

  #pragma unroll
  for (int j = 0; j < 8; ++j) {
    int row = row0 + wave * 16 + 8 * half + j;
    int bb  = row / SEQ;
    int s   = row % SEQ;
    if (part == 2) {
      size_t vb = ((size_t)bb * NHEAD + h) * HDIM * SEQ;
      #pragma unroll
      for (int g = 0; g < 4; ++g)
        vtbuf[vb + (size_t)(g * 16 + ln) * SEQ + s] = (bf16_t)acc[g][j];
    } else {
      size_t hb = (((size_t)bb * NHEAD + h) * SEQ + s) * HDIM;
      bf16_t* dst = (part == 0) ? qbuf : kbuf;
      #pragma unroll
      for (int g = 0; g < 2; ++g) {
        int d0 = g * 16 + ln;                 // < 32
        float fr  = __expf((float)d0 * (-2.0f / (float)HDIM) * LN10000);
        float ang = (float)s * fr;
        float cv = __cosf(ang), sv = __sinf(ang);
        float x1 = acc[g][j], x2 = acc[g + 2][j];
        dst[hb + d0]      = (bf16_t)(x1 * cv - x2 * sv);
        dst[hb + d0 + 32] = (bf16_t)(x2 * cv + x1 * sv);
      }
    }
  }
}

// ---------------------------------------------------------------------------
// Kernel 2: flash attention. grid = (S/128, H, B); wave = 16 q rows.
// K and pre-transposed V tiles double-buffered via async LDS copies.
// ---------------------------------------------------------------------------
__global__ __launch_bounds__(256)
void mha_flash_attn_kernel(const bf16_t* __restrict__ qbuf,   // [B,H,S,HDIM]
                           const bf16_t* __restrict__ kbuf,   // [B,H,S,HDIM]
                           const bf16_t* __restrict__ vtbuf,  // [B,H,HDIM,S]
                           const int* __restrict__ causal_p,
                           bf16_t* __restrict__ obuf /* [NROWS][D_MODEL] bf16 */) {
  __shared__ alignas(16) bf16_t Ksm[2][32 * 64];     // [kv][d], ld=64
  __shared__ alignas(16) bf16_t Vsm[2][64 * 32];     // [d][kv], ld=32
  __shared__ alignas(16) bf16_t Psm[8][16 * 32];     // per-wave P, ld=32

  const int tid  = threadIdx.x;
  const int wave = tid >> 5;
  const int lane = tid & 31;
  const int half = lane >> 4;
  const int ln   = lane & 15;
  const int h    = blockIdx.y;
  const int b    = blockIdx.z;
  const int q0   = blockIdx.x * 128;
  const int qrow = q0 + wave * 16;
  const int causal = causal_p[0];
  const float scale = 0.125f;                 // 1/sqrt(64)
  const float NEG_INF = -__builtin_inff();

  const bf16_t* qbase  = qbuf  + ((size_t)b * NHEAD + h) * SEQ * HDIM;
  const bf16_t* kbase  = kbuf  + ((size_t)b * NHEAD + h) * SEQ * HDIM;
  const bf16_t* vtbase = vtbuf + ((size_t)b * NHEAD + h) * HDIM * SEQ;

  auto issueKV = [&](int kv0, int buf) {
    int e = tid * 8;                               // 2048 elems per tile
    // K tile is a contiguous 4KB block of [kv][d]
    cp_async16(&Ksm[buf][e], kbase + (size_t)kv0 * HDIM + e);
    // Vt tile: 64 rows (d) x 32 cols (kv); contiguous 64B per row
    int d = e >> 5, ck = e & 31;
    cp_async16(&Vsm[buf][e], vtbase + (size_t)d * SEQ + kv0 + ck);
  };

  // Q held in registers: 16x64 = two 16x32 A fragments
  v16bf aq[2];
  #pragma unroll
  for (int ks = 0; ks < 2; ++ks)
    aq[ks] = load_frag(qbase, qrow + ln, HDIM, ks * 32, half);

  v8f o[4] = {};
  float mrow[8], lrow[8];
  #pragma unroll
  for (int j = 0; j < 8; ++j) { mrow[j] = NEG_INF; lrow[j] = 0.0f; }

  const int kvmax = causal ? (q0 + 128) : SEQ;
  const int T = kvmax / 32;

  issueKV(0, 0);
  for (int t = 0; t < T; ++t) {
    const int buf = t & 1;
    const int kv0 = t * 32;
    if (t + 1 < T) { issueKV((t + 1) * 32, buf ^ 1); async_wait<2>(); }
    else           { async_wait<0>(); }
    __syncthreads();

    // S = Q K^T : two 16x16 C tiles (kv cols 0..15 / 16..31)
    v8f s0 = {}, s1 = {};
    #pragma unroll
    for (int ks = 0; ks < 2; ++ks) {
      s0 = wmma_bf16(aq[ks], load_frag(Ksm[buf], ln,      HDIM, ks * 32, half), s0);
      s1 = wmma_bf16(aq[ks], load_frag(Ksm[buf], 16 + ln, HDIM, ks * 32, half), s1);
    }

    float sv0[8], sv1[8];
    #pragma unroll
    for (int j = 0; j < 8; ++j) {
      int qp = qrow + 8 * half + j;
      float a0 = s0[j] * scale;
      float a1 = s1[j] * scale;
      if (causal && (kv0 + ln)      > qp) a0 = NEG_INF;
      if (causal && (kv0 + 16 + ln) > qp) a1 = NEG_INF;
      sv0[j] = a0; sv1[j] = a1;
    }

    #pragma unroll
    for (int j = 0; j < 8; ++j) {
      float mx = fmaxf(sv0[j], sv1[j]);
      #pragma unroll
      for (int msk = 1; msk < 16; msk <<= 1)
        mx = fmaxf(mx, __shfl_xor(mx, msk, 32));
      float mnew  = fmaxf(mrow[j], mx);
      float alpha = __expf(mrow[j] - mnew);
      float p0 = __expf(sv0[j] - mnew);
      float p1 = __expf(sv1[j] - mnew);
      float ss = p0 + p1;
      #pragma unroll
      for (int msk = 1; msk < 16; msk <<= 1)
        ss += __shfl_xor(ss, msk, 32);
      lrow[j] = lrow[j] * alpha + ss;
      mrow[j] = mnew;
      #pragma unroll
      for (int g = 0; g < 4; ++g) o[g][j] *= alpha;
      Psm[wave][(8 * half + j) * 32 + ln]      = (bf16_t)p0;
      Psm[wave][(8 * half + j) * 32 + 16 + ln] = (bf16_t)p1;
    }

    // O += P(16x32) @ V(32x64)
    v16bf ap = load_frag(&Psm[wave][0], ln, 32, 0, half);
    #pragma unroll
    for (int g = 0; g < 4; ++g)
      o[g] = wmma_bf16(ap, load_frag(Vsm[buf], g * 16 + ln, 32, 0, half), o[g]);

    __syncthreads();
  }

  // normalize + store bf16 O in [row][h*64+d] layout for the final GEMM
  #pragma unroll
  for (int j = 0; j < 8; ++j) {
    int qp = qrow + 8 * half + j;
    float inv = 1.0f / lrow[j];
    size_t rbase = ((size_t)b * SEQ + qp) * D_MODEL + h * HDIM;
    #pragma unroll
    for (int g = 0; g < 4; ++g)
      obuf[rbase + g * 16 + ln] = (bf16_t)(o[g][j] * inv);
  }
}

// ---------------------------------------------------------------------------
// Kernel 3: out = O @ w_proj + b_proj (bf16 WMMA, fp32 out)
// ---------------------------------------------------------------------------
__global__ __launch_bounds__(256)
void mha_out_proj_kernel(const bf16_t* __restrict__ obuf,   // [NROWS][D_MODEL]
                         const bf16_t* __restrict__ wpT,    // [D_MODEL][D_MODEL]
                         const float* __restrict__ b_proj,  // [D_MODEL]
                         float* __restrict__ out) {
  __shared__ alignas(16) bf16_t Asm[2][128 * 32];
  __shared__ alignas(16) bf16_t Bsm[2][64 * 32];

  const int tid  = threadIdx.x;
  const int wave = tid >> 5;
  const int lane = tid & 31;
  const int half = lane >> 4;
  const int ln   = lane & 15;
  const int row0 = blockIdx.x * 128;
  const int col0 = blockIdx.y * 64;

  auto issue = [&](int k0, int buf) {
    #pragma unroll
    for (int c = 0; c < 2; ++c) {
      int e = (c * 256 + tid) * 8;
      int r = e >> 5, cc = e & 31;
      cp_async16(&Asm[buf][e], obuf + (size_t)(row0 + r) * D_MODEL + k0 + cc);
    }
    int e = tid * 8;
    int rn = e >> 5, ck = e & 31;
    cp_async16(&Bsm[buf][e], wpT + (size_t)(col0 + rn) * D_MODEL + k0 + ck);
  };

  v8f acc[4] = {};
  const int T = D_MODEL / 32;
  issue(0, 0);
  for (int t = 0; t < T; ++t) {
    int buf = t & 1;
    if (t + 1 < T) { issue((t + 1) * 32, buf ^ 1); async_wait<3>(); }
    else           { async_wait<0>(); }
    __syncthreads();

    v16bf af = load_frag(Asm[buf], wave * 16 + ln, 32, 0, half);
    #pragma unroll
    for (int g = 0; g < 4; ++g)
      acc[g] = wmma_bf16(af, load_frag(Bsm[buf], g * 16 + ln, 32, 0, half), acc[g]);
    __syncthreads();
  }

  #pragma unroll
  for (int j = 0; j < 8; ++j) {
    int row = row0 + wave * 16 + 8 * half + j;
    #pragma unroll
    for (int g = 0; g < 4; ++g) {
      int col = col0 + g * 16 + ln;
      out[(size_t)row * D_MODEL + col] = acc[g][j] + b_proj[col];
    }
  }
}

// ---------------------------------------------------------------------------
extern "C" void kernel_launch(void* const* d_in, const int* in_sizes, int n_in,
                              void* d_out, int out_size, void* d_ws, size_t ws_size,
                              hipStream_t stream) {
  const float* x      = (const float*)d_in[0];
  const float* w_qkv  = (const float*)d_in[1];
  const float* w_proj = (const float*)d_in[2];
  const float* b_proj = (const float*)d_in[3];
  const int*   causal = (const int*)d_in[4];

  const size_t n_x  = (size_t)NROWS * D_MODEL;            // 4,194,304
  const size_t n_wq = (size_t)D_MODEL * 3 * D_MODEL;      // 3,145,728
  const size_t n_wp = (size_t)D_MODEL * D_MODEL;          // 1,048,576
  const size_t n_h  = (size_t)BATCH * NHEAD * SEQ * HDIM; // 4,194,304

  bf16_t* xb    = (bf16_t*)d_ws;
  bf16_t* wqT   = xb  + n_x;
  bf16_t* wpT   = wqT + n_wq;
  bf16_t* qbuf  = wpT + n_wp;
  bf16_t* kbuf  = qbuf + n_h;
  bf16_t* vtbuf = kbuf + n_h;
  bf16_t* obuf  = vtbuf + n_h;

  // one-time precision/layout prep
  mha_cvt_bf16_kernel<<<dim3((unsigned)(n_x / 4 / 256)), 256, 0, stream>>>(x, xb, (int)n_x);
  mha_cvt_transpose_kernel<<<dim3((unsigned)((n_wq + 255) / 256)), 256, 0, stream>>>(
      w_qkv, wqT, D_MODEL, 3 * D_MODEL);
  mha_cvt_transpose_kernel<<<dim3((unsigned)((n_wp + 255) / 256)), 256, 0, stream>>>(
      w_proj, wpT, D_MODEL, D_MODEL);

  mha_qkv_rope_kernel<<<dim3(NROWS / 128, (3 * D_MODEL) / 64), 256, 0, stream>>>(
      xb, wqT, qbuf, kbuf, vtbuf);

  mha_flash_attn_kernel<<<dim3(SEQ / 128, NHEAD, BATCH), 256, 0, stream>>>(
      qbuf, kbuf, vtbuf, causal, obuf);

  mha_out_proj_kernel<<<dim3(NROWS / 128, D_MODEL / 64), 256, 0, stream>>>(
      obuf, wpT, b_proj, (float*)d_out);
}
